// VisionMambaBlock_6511170420826
// MI455X (gfx1250) — matrix operand order repaired and checked by reading, not compiled, not run on hardware
//
#include <hip/hip_runtime.h>
#include <hip/hip_bf16.h>

// ---- problem constants (from reference) ----
#define B_SZ     2
#define L_SZ     1024
#define DM       1024            // d_model
#define DI       2048            // d_inner
#define DS       16              // d_state
#define DC       4               // d_conv
#define DTR      64              // dt_rank
#define XPN      (DTR + 2*DS)    // 96 = x_proj output cols
#define NT       (B_SZ * L_SZ)   // 2048 tokens

typedef __attribute__((ext_vector_type(16))) _Float16     v16h;
typedef __attribute__((ext_vector_type(8)))  _Float16     v8h;
typedef __attribute__((ext_vector_type(8)))  float        v8f;
typedef __attribute__((ext_vector_type(4)))  unsigned int v4u;
typedef __attribute__((ext_vector_type(8)))  unsigned int v8u;

// ---------------- f32 -> f16 convert (weights) ----------------
__global__ void mb_cvt_f16(const float* __restrict__ src, _Float16* __restrict__ dst, int n) {
    int i = blockIdx.x * blockDim.x + threadIdx.x;
    for (; i < n; i += gridDim.x * blockDim.x) dst[i] = (_Float16)src[i];
}

// ---------------- LayerNorm over DM, emit f16 ----------------
__global__ void mb_layernorm(const float* __restrict__ x, const float* __restrict__ g,
                             const float* __restrict__ b, _Float16* __restrict__ out) {
    const int t = blockIdx.x;
    const int tid = threadIdx.x;        // 256 threads
    __shared__ float s1[256], s2[256];
    const float* xp = x + (size_t)t * DM;
    float a0 = 0.f, a1 = 0.f;
    for (int i = tid; i < DM; i += 256) { float v = xp[i]; a0 += v; a1 += v * v; }
    s1[tid] = a0; s2[tid] = a1; __syncthreads();
    for (int s = 128; s > 0; s >>= 1) {
        if (tid < s) { s1[tid] += s1[tid + s]; s2[tid] += s2[tid + s]; }
        __syncthreads();
    }
    const float mu = s1[0] * (1.f / DM);
    const float var = s2[0] * (1.f / DM) - mu * mu;
    const float rs = rsqrtf(var + 1e-5f);
    _Float16* op = out + (size_t)t * DM;
    for (int i = tid; i < DM; i += 256)
        op[i] = (_Float16)((xp[i] - mu) * rs * g[i] + b[i]);
}

// ---- A-fragment load (contiguous K order, same as B operand) ----
// lane holds row m = lane&15; contiguous K = hi*16+[0,16)
template <typename P>
__device__ __forceinline__ v16h mb_load_afrag(P ap) {
    v8h a0 = *(const v8h*)(ap);
    v8h a1 = *(const v8h*)(ap + 8);
    v16h av;
#pragma unroll
    for (int i = 0; i < 8; ++i) { av[i] = a0[i]; av[i + 8] = a1[i]; }
    return av;
}

// ---------------- generic WMMA GEMM (1 tile / wave) ----------------
// C[M,N] = A[M,K] * W[N,K]^T ; used for the small x_proj GEMM (N=96)
__global__ void mb_gemm_f16(const _Float16* __restrict__ A, const _Float16* __restrict__ W,
                            float* __restrict__ C, int M, int N, int K) {
    const int lane = threadIdx.x & 31;
    const int wave = threadIdx.x >> 5;
    const int n0 = (blockIdx.x * 8 + wave) * 16;
    if (n0 >= N) return;
    const int m0 = blockIdx.y * 16;
    const int lr = lane & 15;
    const int hi = lane >> 4;

    const _Float16* ap = A + (size_t)(m0 + lr) * K + hi * 16;
    const _Float16* wp = W + (size_t)(n0 + lr) * K + hi * 16;   // B: contiguous K = hi*16+[0,16)

    v8f acc = {};
    for (int k0 = 0; k0 < K; k0 += 32) {
        v16h av = mb_load_afrag(ap + k0);
        v16h bv = *(const v16h*)(wp + k0);
        acc = __builtin_amdgcn_wmma_f32_16x16x32_f16(false, av, false, bv,
                                                     (short)0, acc, false, false);
    }
    float* cp = C + (size_t)(m0 + hi * 8) * N + n0 + lr;
#pragma unroll
    for (int v = 0; v < 8; ++v) cp[(size_t)v * N] = acc[v];
}

// ---- TDM: issue tensor_load_to_lds of a 64x32 f16 tile (row stride = K elems) ----
// D# per CDNA5 ISA. 2-D tensor -> VADDR2/3 = NULL (omitted).
__device__ __forceinline__ void mb_tdm_load_a64x32(unsigned int lds_addr,
                                                   const _Float16* gsrc, int strideElems) {
    const unsigned long long ga = (unsigned long long)(uintptr_t)gsrc;
    v4u g0;
    g0[0] = 1u;                                            // count=1 (valid descriptor)
    g0[1] = lds_addr;                                      // lds_addr[31:0]
    g0[2] = (unsigned int)ga;                              // global_addr[31:0]
    g0[3] = ((unsigned int)(ga >> 32) & 0x01FFFFFFu)       // global_addr[56:32]
            | (2u << 30);                                  // type=2 ("image")
    v8u g1;
    g1[0] = 1u << 16;                                      // data_size=1 (2 bytes)
    g1[1] = 32u << 16;                                     // tensor_dim0=32 (lo16 @ bits63:48)
    g1[2] = 64u << 16;                                     // tensor_dim1=64 (lo16 @ bits95:80)
    g1[3] = 32u << 16;                                     // tile_dim0=32   (bits127:112)
    g1[4] = 64u;                                           // tile_dim1=64, tile_dim2=0
    g1[5] = (unsigned int)strideElems;                     // tensor_dim0_stride[31:0]
    g1[6] = 0u;                                            // stride hi / dim1_stride
    g1[7] = 0u;
    asm volatile("tensor_load_to_lds %0, %1" : : "s"(g0), "s"(g1) : "memory");
}

// ---------------- register-blocked WMMA GEMM w/ TDM-staged A ----------------
// Wave computes 64x64 (4x4 fragments). Block = 8 waves along N (512 cols) sharing
// the 64-row A tile, which the TDM streams into LDS (double buffered, TENSORcnt).
// Requires M%64==0, N%512==0, K%32==0 and exact grids (barriers inside).
__global__ void mb_gemm_f16_tdm(const _Float16* __restrict__ A, const _Float16* __restrict__ W,
                                float* __restrict__ C, int M, int N, int K) {
    __shared__ _Float16 smA[2][64 * 32];
    const int lane = threadIdx.x & 31;
    const int wave = threadIdx.x >> 5;
    const int n0 = blockIdx.x * 512 + wave * 64;
    const int m0 = blockIdx.y * 64;
    const int lr = lane & 15;
    const int hi = lane >> 4;

    const unsigned int ldsA[2] = {
        (unsigned int)(uintptr_t)&smA[0][0],   // generic addr low 32 bits == LDS offset
        (unsigned int)(uintptr_t)&smA[1][0]
    };
    const _Float16* atile = A + (size_t)m0 * K;          // 64 rows, row stride K
    const _Float16* wp = W + (size_t)(n0 + lr) * K + hi * 16;
    const size_t rowK16 = (size_t)16 * K;
    const int nsteps = K / 32;

    v8f acc[4][4];
#pragma unroll
    for (int i = 0; i < 4; ++i)
#pragma unroll
        for (int j = 0; j < 4; ++j) acc[i][j] = (v8f){};

    if (wave == 0) mb_tdm_load_a64x32(ldsA[0], atile, K);      // prologue: buf 0

    for (int s = 0; s < nsteps; ++s) {
        if (wave == 0) {
            if (s + 1 < nsteps) {
                mb_tdm_load_a64x32(ldsA[(s + 1) & 1], atile + (s + 1) * 32, K);
                __builtin_amdgcn_s_wait_tensorcnt(1);          // tile s complete (in-order)
            } else {
                __builtin_amdgcn_s_wait_tensorcnt(0);
            }
        }
        __syncthreads();                                       // publish LDS tile s

        const _Float16* ab = &smA[s & 1][0] + lr * 32 + hi * 16;
        const int k0 = s * 32;
        __builtin_prefetch(wp + k0 + 512, 0, 1);
        v16h av[4], bv[4];
#pragma unroll
        for (int i = 0; i < 4; ++i) av[i] = mb_load_afrag(ab + i * 16 * 32);   // ds_load_b128 x2
#pragma unroll
        for (int j = 0; j < 4; ++j) bv[j] = *(const v16h*)(wp + rowK16 * j + k0);
#pragma unroll
        for (int i = 0; i < 4; ++i)
#pragma unroll
            for (int j = 0; j < 4; ++j)
                acc[i][j] = __builtin_amdgcn_wmma_f32_16x16x32_f16(
                    false, av[i], false, bv[j], (short)0, acc[i][j], false, false);

        __syncthreads();                                       // done reading buf before reuse
    }

    // C layout: VGPR v -> row (i*16 + v + hi*8), col (j*16 + lr)
#pragma unroll
    for (int i = 0; i < 4; ++i) {
        float* cpi = C + (size_t)(m0 + i * 16 + hi * 8) * N + n0 + lr;
#pragma unroll
        for (int j = 0; j < 4; ++j) {
#pragma unroll
            for (int v = 0; v < 8; ++v)
                cpi[(size_t)v * N + j * 16] = acc[i][j][v];
        }
    }
}

// ---------------- causal depthwise conv (width 4) + SiLU ----------------
__global__ void mb_conv_silu(const float* __restrict__ xz, const float* __restrict__ cw,
                             const float* __restrict__ cb, float* __restrict__ xh32,
                             _Float16* __restrict__ xh16) {
    size_t idx = (size_t)blockIdx.x * 256 + threadIdx.x;
    if (idx >= (size_t)NT * DI) return;
    const int d = (int)(idx % DI);
    const size_t t = idx / DI;
    const int l = (int)(t % L_SZ);
    const size_t b = t / L_SZ;
    float acc = cb[d];
#pragma unroll
    for (int j = 0; j < DC; ++j) {
        int ll = l - (DC - 1) + j;
        if (ll >= 0)
            acc += xz[((b * L_SZ + ll) * (size_t)(2 * DI)) + d] * cw[d * DC + j];
    }
    const float s = acc / (1.f + expf(-acc));   // SiLU
    xh32[idx] = s;
    xh16[idx] = (_Float16)s;
}

// ---------------- extract dt columns of x_dbl as f16 ----------------
__global__ void mb_pack_dt(const float* __restrict__ xdbl, _Float16* __restrict__ dt16) {
    size_t idx = (size_t)blockIdx.x * 256 + threadIdx.x;
    if (idx >= (size_t)NT * DTR) return;
    const size_t t = idx / DTR;
    const int r = (int)(idx % DTR);
    dt16[idx] = (_Float16)xdbl[t * XPN + r];
}

// ---------------- selective scan over L ----------------
// one thread per (b,d) channel; Bc/Cc are block-uniform -> scalar loads
__global__ void mb_scan(const float* __restrict__ draw,   // [NT, DI] pre-softplus delta
                        const float* __restrict__ xdbl,   // [NT, 96] (B at 64..80, C at 80..96)
                        const float* __restrict__ xh32,   // [NT, DI]
                        const float* __restrict__ A_log,  // [DI, DS]
                        const float* __restrict__ dtb,    // [DI]
                        const float* __restrict__ Dp,     // [DI]
                        float* __restrict__ ys) {         // [NT, DI] out: y + D*xh
    const int d = blockIdx.x * 256 + threadIdx.x;
    const int b = blockIdx.y;

    float Aa[DS], h[DS];
#pragma unroll
    for (int n = 0; n < DS; ++n) {
        Aa[n] = -expf(A_log[(size_t)d * DS + n]);
        h[n] = 0.f;
    }
    const float bias = dtb[d];
    const float Dd = Dp[d];

    for (int l = 0; l < L_SZ; ++l) {
        const size_t t = (size_t)b * L_SZ + l;
        const float* bc = xdbl + t * XPN + DTR;         // uniform across block
        float dlt = draw[t * DI + d] + bias;
        dlt = (dlt > 20.f) ? dlt : log1pf(expf(dlt));   // softplus
        const float xv = xh32[t * DI + d];
        const float dx = dlt * xv;
        float y = 0.f;
#pragma unroll
        for (int n = 0; n < DS; ++n) {
            h[n] = expf(dlt * Aa[n]) * h[n] + bc[n] * dx;
            y += h[n] * bc[DS + n];
        }
        ys[t * DI + d] = y + Dd * xv;
    }
}

// ---------------- gate: y *= silu(z), emit f16 ----------------
__global__ void mb_gate(const float* __restrict__ ys, const float* __restrict__ xz,
                        _Float16* __restrict__ y16) {
    size_t idx = (size_t)blockIdx.x * 256 + threadIdx.x;
    if (idx >= (size_t)NT * DI) return;
    const size_t t = idx / DI;
    const int d = (int)(idx % DI);
    const float z = xz[t * (size_t)(2 * DI) + DI + d];
    y16[idx] = (_Float16)(ys[idx] * (z / (1.f + expf(-z))));
}

// ---------------- residual add ----------------
__global__ void mb_residual(const float* __restrict__ yproj, const float* __restrict__ x,
                            float* __restrict__ out) {
    size_t idx = (size_t)blockIdx.x * 256 + threadIdx.x;
    if (idx >= (size_t)NT * DM) return;
    out[idx] = yproj[idx] + x[idx];
}

extern "C" void kernel_launch(void* const* d_in, const int* in_sizes, int n_in,
                              void* d_out, int out_size, void* d_ws, size_t ws_size,
                              hipStream_t stream) {
    (void)in_sizes; (void)n_in; (void)out_size; (void)ws_size;
    const float* x       = (const float*)d_in[0];
    const float* ln_g    = (const float*)d_in[1];
    const float* ln_b    = (const float*)d_in[2];
    const float* in_w    = (const float*)d_in[3];   // [2*DI, DM]
    const float* conv_w  = (const float*)d_in[4];   // [DI, DC]
    const float* conv_b  = (const float*)d_in[5];   // [DI]
    const float* xp_w    = (const float*)d_in[6];   // [96, DI]
    const float* dtp_w   = (const float*)d_in[7];   // [DI, DTR]
    const float* dtp_b   = (const float*)d_in[8];   // [DI]
    const float* A_log   = (const float*)d_in[9];   // [DI, DS]
    const float* Dp      = (const float*)d_in[10];  // [DI]
    const float* out_w   = (const float*)d_in[11];  // [DM, DI]
    float* out = (float*)d_out;

    // ---- workspace sub-allocation (256B aligned) ----
    char* ws = (char*)d_ws;
    size_t off = 0;
    auto alloc = [&](size_t bytes) -> char* {
        char* p = ws + off;
        off = (off + bytes + 255) & ~(size_t)255;
        return p;
    };
    _Float16* w16_in  = (_Float16*)alloc((size_t)2 * DI * DM * 2);
    _Float16* w16_xp  = (_Float16*)alloc((size_t)XPN * DI * 2);
    _Float16* w16_dt  = (_Float16*)alloc((size_t)DI * DTR * 2);
    _Float16* w16_out = (_Float16*)alloc((size_t)DM * DI * 2);
    _Float16* xn16    = (_Float16*)alloc((size_t)NT * DM * 2);
    float*    xz      = (float*)   alloc((size_t)NT * 2 * DI * 4);
    float*    xh32    = (float*)   alloc((size_t)NT * DI * 4);
    _Float16* xh16    = (_Float16*)alloc((size_t)NT * DI * 2);
    float*    xdbl    = (float*)   alloc((size_t)NT * XPN * 4);
    _Float16* dt16    = (_Float16*)alloc((size_t)NT * DTR * 2);
    float*    draw    = (float*)   alloc((size_t)NT * DI * 4);
    float*    ys      = (float*)   alloc((size_t)NT * DI * 4);
    _Float16* y16     = (_Float16*)alloc((size_t)NT * DI * 2);
    float*    yproj   = (float*)   alloc((size_t)NT * DM * 4);

    // 1) weight conversion to f16
    {
        int n1 = 2 * DI * DM, n2 = XPN * DI, n3 = DI * DTR, n4 = DM * DI;
        mb_cvt_f16<<<(n1 + 255) / 256, 256, 0, stream>>>(in_w,  w16_in,  n1);
        mb_cvt_f16<<<(n2 + 255) / 256, 256, 0, stream>>>(xp_w,  w16_xp,  n2);
        mb_cvt_f16<<<(n3 + 255) / 256, 256, 0, stream>>>(dtp_w, w16_dt,  n3);
        mb_cvt_f16<<<(n4 + 255) / 256, 256, 0, stream>>>(out_w, w16_out, n4);
    }
    // 2) LayerNorm -> xn16
    mb_layernorm<<<NT, 256, 0, stream>>>(x, ln_g, ln_b, xn16);
    // 3) in_proj GEMM (TDM-staged): [NT,DM] x [2*DI,DM]^T -> xz [NT, 2*DI]
    mb_gemm_f16_tdm<<<dim3(2 * DI / 512, NT / 64), 256, 0, stream>>>(
        xn16, w16_in, xz, NT, 2 * DI, DM);
    // 4) causal depthwise conv + SiLU -> xh
    mb_conv_silu<<<((size_t)NT * DI + 255) / 256, 256, 0, stream>>>(xz, conv_w, conv_b, xh32, xh16);
    // 5) x_proj GEMM (small N): [NT,DI] x [96,DI]^T -> xdbl [NT,96]
    mb_gemm_f16<<<dim3((XPN / 16 + 7) / 8, NT / 16), 256, 0, stream>>>(
        xh16, w16_xp, xdbl, NT, XPN, DI);
    // 6) extract dt as f16
    mb_pack_dt<<<((size_t)NT * DTR + 255) / 256, 256, 0, stream>>>(xdbl, dt16);
    // 7) dt_proj GEMM (TDM-staged): [NT,DTR] x [DI,DTR]^T -> draw [NT,DI]
    mb_gemm_f16_tdm<<<dim3(DI / 512, NT / 64), 256, 0, stream>>>(
        dt16, w16_dt, draw, NT, DI, DTR);
    // 8) selective scan (softplus + SSM recurrence + D skip)
    mb_scan<<<dim3(DI / 256, B_SZ), 256, 0, stream>>>(draw, xdbl, xh32, A_log, dtp_b, Dp, ys);
    // 9) gate with silu(z)
    mb_gate<<<((size_t)NT * DI + 255) / 256, 256, 0, stream>>>(ys, xz, y16);
    // 10) out_proj GEMM (TDM-staged): [NT,DI] x [DM,DI]^T -> yproj [NT,DM]
    mb_gemm_f16_tdm<<<dim3(DM / 512, NT / 64), 256, 0, stream>>>(
        y16, w16_out, yproj, NT, DM, DI);
    // 11) residual add -> d_out
    mb_residual<<<((size_t)NT * DM + 255) / 256, 256, 0, stream>>>(yproj, x, out);
}
